// SlidingWindowAttention_65171833749623
// MI455X (gfx1250) — matrix operand order repaired and checked by reading, not compiled, not run on hardware
//
#include <hip/hip_runtime.h>
#include <math.h>

typedef __bf16 bf16_t;
typedef bf16_t v16bf __attribute__((ext_vector_type(16)));
typedef bf16_t v8bf  __attribute__((ext_vector_type(8)));
typedef float  v8f   __attribute__((ext_vector_type(8)));

#define WIN    512
#define HDIM   64
#define NWIN   16
#define NHEAD  16
#define NBATCH 2
#define SEQ    8192
#define STRIP  64
#define NSTRIP 8
#define FADEN  64
#define NWAVES 16

static __device__ __forceinline__ bf16_t f2bf(float f) {
  unsigned u = __builtin_bit_cast(unsigned, f);
  unsigned r = u + 0x7FFFu + ((u >> 16) & 1u);   // round-to-nearest-even
  unsigned short hh = (unsigned short)(r >> 16);
  return __builtin_bit_cast(bf16_t, hh);
}

// inv_freq[j] = 10000^(-j/32) = exp2(-j * log2(10000)/32)
static __device__ __forceinline__ float ropeInvFreq(int j) {
  return exp2f(-(float)j * (13.287712379549449f / 32.0f));
}

static __device__ __forceinline__ v8f wmma_bf16(v16bf a, v16bf b, v8f c) {
  // D = A(16x32 bf16) * B(32x16 bf16) + C(16x16 f32)
  return __builtin_amdgcn_wmma_f32_16x16x32_bf16(
      /*neg_a=*/false, a, /*neg_b=*/false, b,
      /*c_mod=*/(short)0, c, /*reuse_a=*/false, /*reuse_b=*/false);
}

// xor-butterfly within each 16-lane row via v_permlane16_b32 (pure VALU, no DS)
static __device__ __forceinline__ float permxor16(float x, int sel0, int sel1) {
  int v = __builtin_bit_cast(int, x);
  int r = __builtin_amdgcn_permlane16(v, v, sel0, sel1, false, false);
  return __builtin_bit_cast(float, r);
}
static __device__ __forceinline__ float row16_max(float t) {
  t = fmaxf(t, permxor16(t, 0x67452301, 0xEFCDAB89));  // xor 1
  t = fmaxf(t, permxor16(t, 0x54761032, 0xDCFE98BA));  // xor 2
  t = fmaxf(t, permxor16(t, 0x32107654, 0xBA98FEDC));  // xor 4
  t = fmaxf(t, permxor16(t, 0xFEDCBA98, 0x76543210));  // xor 8
  return t;
}
static __device__ __forceinline__ float row16_sum(float t) {
  t += permxor16(t, 0x67452301, 0xEFCDAB89);
  t += permxor16(t, 0x54761032, 0xDCFE98BA);
  t += permxor16(t, 0x32107654, 0xBA98FEDC);
  t += permxor16(t, 0xFEDCBA98, 0x76543210);
  return t;
}

__global__ __launch_bounds__(512)
void swa_fused_kernel(const float* __restrict__ q,
                      const float* __restrict__ k,
                      const float* __restrict__ v,
                      float* __restrict__ out)
{
  __shared__ bf16_t ldsK[STRIP][HDIM];      // RoPE'd keys, bf16, [key][d]   (8 KB)
  __shared__ bf16_t ldsVT[HDIM][STRIP];     // V transposed,   [d][key]      (8 KB)
  __shared__ bf16_t ldsP[NWAVES][16][32];   // per-wave P transpose scratch (16 KB)

  const int tid  = threadIdx.x;
  const int lane = tid & 31;
  const int wv   = tid >> 5;
  const int ln16 = lane & 15;
  const int hi   = (lane >= 16) ? 1 : 0;

  const int blk = blockIdx.x;
  const int n = blk % NWIN;
  const int h = (blk / NWIN) % NHEAD;
  const int b = blk / (NWIN * NHEAD);

  const size_t winOff = (((size_t)(b * NHEAD + h)) * SEQ + (size_t)n * WIN) * HDIM;

  // ---------- persistent RoPE'd Q fragments (A-layout, bf16) ----------
  // 1/sqrt(64) folded into Q here so scores come out of WMMA pre-scaled.
  v16bf qA[2][2];
  const int qbase = wv * 32;
  for (int mt = 0; mt < 2; ++mt) {
    const int row = qbase + mt * 16 + ln16;
    const float pos = (float)(n * WIN + row);        // position_ids == arange
    const float* qrow = q + winOff + (size_t)row * HDIM;
    for (int c = 0; c < 2; ++c) {
      v16bf a;
      for (int i = 0; i < 16; ++i) {
        int d = 32 * c + 16 * hi + i;
        float ang = pos * ropeInvFreq(d & 31);
        float sn, cs; __sincosf(ang, &sn, &cs);
        float x  = qrow[d];
        float xp = qrow[d ^ 32];
        float rot = (d < 32) ? -xp : xp;
        a[i] = f2bf((x * cs + rot * sn) * 0.125f);
      }
      qA[mt][c] = a;
    }
  }

  // ---------- flash-attention running state ----------
  v8f   acc[2][4];
  float mrow[2][8], lrow[2][8];
  for (int mt = 0; mt < 2; ++mt) {
    for (int dc = 0; dc < 4; ++dc)
      for (int e = 0; e < 8; ++e) acc[mt][dc][e] = 0.0f;
    for (int r = 0; r < 8; ++r) { mrow[mt][r] = -1.0e30f; lrow[mt][r] = 0.0f; }
  }

  for (int s = 0; s < NSTRIP; ++s) {
    // ---------- cooperative staging of one 64-key strip ----------
    if (tid < 256) {
      // RoPE'd K -> ldsK[key][d] (each thread: 8 d-pairs (d, d+32), same angle)
      const int key = tid >> 2, qt = tid & 3, d0 = 8 * qt;
      const int kg  = s * STRIP + key;
      const float pos = (float)(n * WIN + kg);
      const float* krow = k + winOff + (size_t)kg * HDIM;
      const float4* k4 = (const float4*)krow;
      float4 x0 = k4[d0 / 4],        x1 = k4[d0 / 4 + 1];
      float4 y0 = k4[(d0 + 32) / 4], y1 = k4[(d0 + 32) / 4 + 1];
      float xs[8] = {x0.x,x0.y,x0.z,x0.w,x1.x,x1.y,x1.z,x1.w};
      float ys[8] = {y0.x,y0.y,y0.z,y0.w,y1.x,y1.y,y1.z,y1.w};
      for (int j = 0; j < 8; ++j) {
        int d = d0 + j;                      // d < 32
        float ang = pos * ropeInvFreq(d);
        float sn, cs; __sincosf(ang, &sn, &cs);
        ldsK[key][d]      = f2bf(xs[j] * cs - ys[j] * sn);
        ldsK[key][d + 32] = f2bf(ys[j] * cs + xs[j] * sn);
      }
      // prefetch next strip's K row (global_prefetch_b8)
      if (s + 1 < NSTRIP)
        __builtin_prefetch(krow + (size_t)STRIP * HDIM + d0, 0, 1);
    } else {
      // V -> ldsVT[d][key] (transposed, bf16)
      const int t2 = tid - 256;
      const int key = t2 >> 2, qt = t2 & 3, d0 = 16 * qt;
      const int kg  = s * STRIP + key;
      const float* vrow = v + winOff + (size_t)kg * HDIM;
      const float4* v4 = (const float4*)vrow;
      for (int g = 0; g < 4; ++g) {
        float4 val = v4[d0 / 4 + g];
        ldsVT[d0 + 4 * g + 0][key] = f2bf(val.x);
        ldsVT[d0 + 4 * g + 1][key] = f2bf(val.y);
        ldsVT[d0 + 4 * g + 2][key] = f2bf(val.z);
        ldsVT[d0 + 4 * g + 3][key] = f2bf(val.w);
      }
      if (s + 1 < NSTRIP)
        __builtin_prefetch(vrow + (size_t)STRIP * HDIM + d0, 0, 1);
    }
    __syncthreads();

    // ---------- two 32-key substeps per strip ----------
    for (int ss = 0; ss < 2; ++ss) {
      const int kb = 32 * ss;

      // B-frags (contiguous k order used here, same as A: lanes 0-15 K=0-15, lanes 16-31 K=16-31, N = lane%16)
      v16bf Bk[2][2], Bv[4];
      for (int j = 0; j < 2; ++j)
        for (int c = 0; c < 2; ++c)
          Bk[j][c] = *(const v16bf*)&ldsK[kb + 16 * j + ln16][32 * c + 16 * hi];
      for (int dc = 0; dc < 4; ++dc)
        Bv[dc] = *(const v16bf*)&ldsVT[dc * 16 + ln16][kb + 16 * hi];

      for (int mt = 0; mt < 2; ++mt) {
        // scores: S = (Q/8)(16x64) . K^T -> two 16x16 tiles (D split into 2 K-chunks)
        v8f S0, S1;
        for (int e = 0; e < 8; ++e) { S0[e] = 0.0f; S1[e] = 0.0f; }
        S0 = wmma_bf16(qA[mt][0], Bk[0][0], S0);
        S0 = wmma_bf16(qA[mt][1], Bk[0][1], S0);
        S1 = wmma_bf16(qA[mt][0], Bk[1][0], S1);
        S1 = wmma_bf16(qA[mt][1], Bk[1][1], S1);

        // online softmax (row = VGPR index + 8*half; 16 lanes of a half = one row)
        float p0[8], p1[8];
        for (int r = 0; r < 8; ++r) {
          float s0 = S0[r];
          float s1 = S1[r];
          float t = row16_max(fmaxf(s0, s1));
          float mold = mrow[mt][r];
          float mnew = fmaxf(mold, t);
          float corr = __expf(mold - mnew);
          mrow[mt][r] = mnew;
          float e0 = __expf(s0 - mnew);
          float e1 = __expf(s1 - mnew);
          float rs = row16_sum(e0 + e1);
          lrow[mt][r] = lrow[mt][r] * corr + rs;
          for (int dc = 0; dc < 4; ++dc) acc[mt][dc][r] *= corr;
          p0[r] = e0; p1[r] = e1;
        }

        // P: C-layout -> A-layout via wave-private LDS bounce (DS is in-order per wave)
        for (int r = 0; r < 8; ++r) {
          ldsP[wv][r + 8 * hi][ln16]      = f2bf(p0[r]);
          ldsP[wv][r + 8 * hi][16 + ln16] = f2bf(p1[r]);
        }
        asm volatile("s_wait_dscnt 0" ::: "memory");
        union { v16bf v16; v8bf h[2]; } u;
        u.h[0] = *(const v8bf*)&ldsP[wv][ln16][16 * hi];
        u.h[1] = *(const v8bf*)&ldsP[wv][ln16][16 * hi + 8];
        v16bf pA = u.v16;

        // O += P(16x32) . V(32x64), 4 d-chunks
        for (int dc = 0; dc < 4; ++dc)
          acc[mt][dc] = wmma_bf16(pA, Bv[dc], acc[mt][dc]);
      }
    }
    __syncthreads();
  }

  // ---------- finalize: 1/l and fade scale, write fp32 ----------
  const bool edge = (n == 0) || (n == NWIN - 1);
  for (int mt = 0; mt < 2; ++mt) {
    for (int r = 0; r < 8; ++r) {
      const int row = qbase + mt * 16 + r + 8 * hi;
      float fs = 1.0f;
      if (!edge) {
        float wf = 1.0f;
        if (row < FADEN)             wf = (float)row * (1.0f / 63.0f);
        else if (row >= WIN - FADEN) wf = (float)(WIN - 1 - row) * (1.0f / 63.0f);
        fs = wf / (wf + 1e-8f);
      }
      const float scl = fs / lrow[mt][r];
      float* orow = out + winOff + (size_t)row * HDIM;
      for (int dc = 0; dc < 4; ++dc)
        orow[dc * 16 + ln16] = acc[mt][dc][r] * scl;
    }
  }
}

extern "C" void kernel_launch(void* const* d_in, const int* in_sizes, int n_in,
                              void* d_out, int out_size, void* d_ws, size_t ws_size,
                              hipStream_t stream) {
  (void)in_sizes; (void)n_in; (void)d_ws; (void)ws_size; (void)out_size;
  const float* q = (const float*)d_in[0];
  const float* k = (const float*)d_in[1];
  const float* v = (const float*)d_in[2];
  // d_in[3] = position_ids (arange(L)); positions are recomputed analytically.
  float* out = (float*)d_out;
  dim3 grid(NBATCH * NHEAD * NWIN);   // 512 windows
  dim3 block(512);                    // 16 wave32 waves per workgroup
  swa_fused_kernel<<<grid, block, 0, stream>>>(q, k, v, out);
}
